// Decoder_47253230191357
// MI455X (gfx1250) — compile-verified, matrix operand order checked by reading, not run on hardware
//
#include <hip/hip_runtime.h>
#include <hip/hip_bf16.h>
#include <math.h>

// ---------------- CDNA5 WMMA types ----------------
typedef __attribute__((ext_vector_type(16))) __bf16 bfvec16;
typedef __attribute__((ext_vector_type(8)))  float  v8f;

// ---------------- problem constants ----------------
#define B_      32
#define T_ENC   256
#define ENC_D   512
#define T_DEC   500
#define MEL     80
#define R_      2
#define NSTEPS  250      // T_DEC / R
#define PRE     256
#define ATTN_D  128
#define RNN     1024
#define NGATE   4096
#define KA      1792     // PRE + ENC_D + RNN   (attention LSTM combined K)
#define KD      2560     // RNN + ENC_D + RNN   (decoder  LSTM combined K)
#define NKT_A   56       // KA / 32
#define NKT_D   80       // KD / 32
#define FILT    32
#define KSZ     31
#define PADC    15
#define PROJ_K  1536     // RNN + ENC_D

// ---------------- workspace layout (bytes) ----------------
// bf16 stored as unsigned short (manual RNE f32->bf16), loaded as bfvec16.
static const size_t OFF_WA    = 0;
static const size_t SZ_WA     = (size_t)256 * NKT_A * 32 * 16 * 2;   // 14,680,064
static const size_t OFF_WD    = OFF_WA + SZ_WA;
static const size_t SZ_WD     = (size_t)256 * NKT_D * 32 * 16 * 2;   // 20,971,520
static const size_t OFF_PM    = OFF_WD + SZ_WD;
static const size_t SZ_PM     = (size_t)B_ * T_ENC * ATTN_D * 4;     //  4,194,304
static const size_t OFF_AA    = OFF_PM + SZ_PM;
static const size_t SZ_AA     = (size_t)2 * NKT_A * 32 * 16 * 2;     //    114,688
static const size_t OFF_AD    = OFF_AA + SZ_AA;
static const size_t SZ_AD     = (size_t)2 * NKT_D * 32 * 16 * 2;     //    163,840
static const size_t OFF_G     = OFF_AD + SZ_AD;
static const size_t SZ_G      = (size_t)B_ * NGATE * 4;              //    524,288
static const size_t OFF_STATE = OFF_G + SZ_G;
// state floats: attn_h | attn_c | dec_h | dec_c | ctx | cum (contiguous)
static const int N_STATE = B_*RNN*4 + B_*ENC_D + B_*T_ENC;           //    155,648

// ---------------- helpers ----------------
__device__ __forceinline__ float sigmoidf_(float x) { return 1.0f / (1.0f + __expf(-x)); }

__device__ __forceinline__ unsigned short f32_to_bf16(float f) {
    unsigned int u = __float_as_uint(f);
    u += 0x7FFFu + ((u >> 16) & 1u);   // round-to-nearest-even
    return (unsigned short)(u >> 16);
}

// K offset inside a 16x32 bf16 fragment for dword v (0..7) and lane half.
// CDNA5 ISA §7.12.2: VGPR v<4 -> K = 2v (+8 for lanes 16..31),
//                         VGPR v>=4 -> K = 16+2(v-4) (+8 for lanes 16..31).
__device__ __forceinline__ int frag_k(int v, int khalf) {
    return ((v < 4) ? 2 * v : 2 * v + 8) + 8 * khalf;
}

// ---------------- init: pack [Wih | Whh] (f32 row-major) into bf16 B-fragments ----
// Layout: [nt][kt][lane][16 bf16]; lane holds column n = 16*nt + (lane&15),
// K striped per frag_k. One thread per packed dword (2 elements).
__global__ void pack_w_kernel(const float* __restrict__ Wih, int Kih,
                              const float* __restrict__ Whh, int Khh,
                              unsigned short* __restrict__ out, int nkt) {
    int id = blockIdx.x * blockDim.x + threadIdx.x;
    int total = 256 * nkt * 32 * 8;
    if (id >= total) return;
    int v     = id & 7;
    int lane  = (id >> 3) & 31;
    int kt    = (id >> 8) % nkt;
    int nt    = (id >> 8) / nkt;
    int n     = nt * 16 + (lane & 15);
    int khalf = lane >> 4;
    int k0    = kt * 32 + frag_k(v, khalf);
    float w0, w1;
    {
        int k = k0;
        w0 = (k < Kih) ? Wih[(size_t)n * Kih + k] : Whh[(size_t)n * Khh + (k - Kih)];
        k = k0 + 1;
        w1 = (k < Kih) ? Wih[(size_t)n * Kih + k] : Whh[(size_t)n * Khh + (k - Kih)];
    }
    unsigned short* dst = out + (((size_t)(nt * nkt + kt) * 32 + lane) * 16 + 2 * v);
    dst[0] = f32_to_bf16(w0);
    dst[1] = f32_to_bf16(w1);
}

// ---------------- init: processed_memory = enc @ Wm^T  (one-time, fp32) -------
__global__ void pm_kernel(const float* __restrict__ enc, const float* __restrict__ Wm,
                          float* __restrict__ pm) {
    int id = blockIdx.x * blockDim.x + threadIdx.x;
    if (id >= B_ * T_ENC * ATTN_D) return;
    int d  = id & (ATTN_D - 1);
    int bt = id >> 7;
    const float* e = enc + (size_t)bt * ENC_D;
    const float* w = Wm + (size_t)d * ENC_D;
    float s = 0.f;
    #pragma unroll 8
    for (int k = 0; k < ENC_D; ++k) s += e[k] * w[k];
    pm[id] = s;
}

__global__ void zero_kernel(float* p, int n) {
    int i = blockIdx.x * blockDim.x + threadIdx.x;
    if (i < n) p[i] = 0.f;
}

// ---------------- per-step S1: prenet + pack A-fragments for attention LSTM ---
// u1 = [prenet_x(256) | ctx(512) | attn_h(1024)], one block per batch row.
__global__ void prenet_pack_kernel(const float* __restrict__ inputs,
                                   const float* __restrict__ W1, const float* __restrict__ b1,
                                   const float* __restrict__ W2, const float* __restrict__ b2,
                                   const float* __restrict__ ctx, const float* __restrict__ attn_h,
                                   unsigned short* __restrict__ apack, int step) {
    __shared__ float sfr[MEL];
    __shared__ float sx1[PRE];
    __shared__ float sx2[PRE];
    int b = blockIdx.x, tid = threadIdx.x;
    if (tid < MEL) {
        float f = 0.f;
        if (step > 0) f = inputs[((size_t)b * T_DEC + (2 * step - 1)) * MEL + tid];
        sfr[tid] = f;
    }
    __syncthreads();
    {
        float s = b1[tid];
        const float* w = W1 + (size_t)tid * MEL;
        #pragma unroll 8
        for (int m = 0; m < MEL; ++m) s += sfr[m] * w[m];
        sx1[tid] = fmaxf(s, 0.f);
    }
    __syncthreads();
    {
        float s = b2[tid];
        const float* w = W2 + (size_t)tid * PRE;
        #pragma unroll 8
        for (int m = 0; m < PRE; ++m) s += sx1[m] * w[m];
        sx2[tid] = fmaxf(s, 0.f);
    }
    __syncthreads();
    // pack this row's A-fragment lanes (lane = b&15 and (b&15)+16 of M-tile b>>4)
    int mt = b >> 4, lanebase = b & 15;
    const float* ctxb = ctx + (size_t)b * ENC_D;
    const float* hb   = attn_h + (size_t)b * RNN;
    for (int idx = tid; idx < NKT_A * 16; idx += blockDim.x) {
        int kt = idx >> 4, khalf = (idx >> 3) & 1, v = idx & 7;
        int lane = lanebase + 16 * khalf;
        int k0 = kt * 32 + frag_k(v, khalf);
        float w0, w1;
        {
            int k = k0;
            w0 = (k < PRE) ? sx2[k] : (k < PRE + ENC_D) ? ctxb[k - PRE] : hb[k - PRE - ENC_D];
            k = k0 + 1;
            w1 = (k < PRE) ? sx2[k] : (k < PRE + ENC_D) ? ctxb[k - PRE] : hb[k - PRE - ENC_D];
        }
        unsigned short* dst = apack + (((size_t)(mt * NKT_A + kt) * 32 + lane) * 16 + 2 * v);
        dst[0] = f32_to_bf16(w0);
        dst[1] = f32_to_bf16(w1);
    }
}

// ---------------- WMMA GEMM: gates(32x4096) = U(32xK) @ W^T(Kx4096) ----------
// 32 blocks x 8 waves; block stages the entire A-pack (112/160 KB) in LDS once
// (fits CDNA5's 320 KB WGP LDS), so A global traffic is per-block instead of
// per-wave (8x reduction).  Inner loop: 4x ds_load_b128 + 2x global_load_b128
// + 2x v_wmma_f32_16x16x32_bf16.
__global__ __launch_bounds__(256) void gemm_wmma_kernel(
        const unsigned short* __restrict__ apack,
        const unsigned short* __restrict__ wpack,
        float* __restrict__ gates, int nkt) {
    extern __shared__ unsigned short sA[];   // 2 * nkt * 512 bf16 values
    int tid = threadIdx.x;
    // cooperative copy of the whole A-pack into LDS (16B chunks)
    {
        int n16 = (2 * nkt * 512) / 8;       // number of 16-byte chunks
        const uint4* src = reinterpret_cast<const uint4*>(apack);
        uint4* dst = reinterpret_cast<uint4*>(sA);
        for (int i = tid; i < n16; i += 256) dst[i] = src[i];
    }
    __syncthreads();
    int wave = tid >> 5;
    int lane = tid & 31;
    int nt = blockIdx.x * 8 + wave;          // 0..255 N-tiles
    v8f acc0 = {}, acc1 = {};
    const unsigned short* aptr0 = sA + (size_t)lane * 16;                  // mt=0
    const unsigned short* aptr1 = sA + ((size_t)nkt * 32 + lane) * 16;     // mt=1
    const unsigned short* bptr  = wpack + (((size_t)nt * nkt * 32) + lane) * 16;
    for (int kt = 0; kt < nkt; ++kt) {
        bfvec16 a0 = *reinterpret_cast<const bfvec16*>(aptr0 + (size_t)kt * 512);
        bfvec16 a1 = *reinterpret_cast<const bfvec16*>(aptr1 + (size_t)kt * 512);
        bfvec16 bb = *reinterpret_cast<const bfvec16*>(bptr  + (size_t)kt * 512);
        acc0 = __builtin_amdgcn_wmma_f32_16x16x32_bf16(false, a0, false, bb,
                                                       (short)0, acc0, false, false);
        acc1 = __builtin_amdgcn_wmma_f32_16x16x32_bf16(false, a1, false, bb,
                                                       (short)0, acc1, false, false);
    }
    // C/D layout: lanes 0-15 -> M=v, lanes 16-31 -> M=8+v; N = lane&15
    int ncol = nt * 16 + (lane & 15);
    int mrow = (lane >> 4) * 8;
    #pragma unroll
    for (int v = 0; v < 8; ++v) {
        gates[(size_t)(mrow + v) * NGATE + ncol]      = acc0[v];
        gates[(size_t)(16 + mrow + v) * NGATE + ncol] = acc1[v];
    }
}

// ---------------- S3: attention LSTM cell (elementwise) ----------------------
__global__ void lstm_cell_kernel(const float* __restrict__ gates,
                                 const float* __restrict__ bih, const float* __restrict__ bhh,
                                 float* __restrict__ h, float* __restrict__ c) {
    int b = blockIdx.x, tid = threadIdx.x;
    const float* g = gates + (size_t)b * NGATE;
    for (int j = tid; j < RNN; j += blockDim.x) {
        float gi = g[j]        + bih[j]        + bhh[j];
        float gf = g[j + 1024] + bih[j + 1024] + bhh[j + 1024];
        float gg = g[j + 2048] + bih[j + 2048] + bhh[j + 2048];
        float go = g[j + 3072] + bih[j + 3072] + bhh[j + 3072];
        float cc = sigmoidf_(gf) * c[(size_t)b * RNN + j] + sigmoidf_(gi) * tanhf(gg);
        float hh = sigmoidf_(go) * tanhf(cc);
        c[(size_t)b * RNN + j] = cc;
        h[(size_t)b * RNN + j] = hh;
    }
}

// ---------------- S4: fused location-sensitive attention + pack decoder A ----
// One block (256 threads) per batch row. Produces ctx, cum, align (-> d_out),
// then packs u2 = [attn_h | ctx | dec_h_prev] into decoder A-fragments.
__global__ void attention_kernel(const float* __restrict__ enc, const float* __restrict__ pm,
                                 const int* __restrict__ mlen,
                                 const float* __restrict__ Wq,
                                 const float* __restrict__ conv_w, const float* __restrict__ conv_b,
                                 const float* __restrict__ WL, const float* __restrict__ vvec,
                                 const float* __restrict__ attn_h, const float* __restrict__ dec_h,
                                 float* __restrict__ ctx, float* __restrict__ cum,
                                 float* __restrict__ out_attn,
                                 unsigned short* __restrict__ apack_d, int step) {
    __shared__ float sh_h[RNN];
    __shared__ float sh_pq[ATTN_D];
    __shared__ float sh_cum[T_ENC + 2 * PADC];
    __shared__ float sh_e[T_ENC];
    __shared__ float sh_ctx[ENC_D];
    __shared__ float sred[256];
    int b = blockIdx.x, tid = threadIdx.x;
    for (int j = tid; j < RNN; j += 256) sh_h[j] = attn_h[(size_t)b * RNN + j];
    for (int j = tid; j < T_ENC + 2 * PADC; j += 256) {
        int t = j - PADC;
        sh_cum[j] = (t >= 0 && t < T_ENC) ? cum[(size_t)b * T_ENC + t] : 0.f;
    }
    __syncthreads();
    if (tid < ATTN_D) {   // pq = attn_h @ Wq^T
        float s = 0.f;
        const float* w = Wq + (size_t)tid * RNN;
        #pragma unroll 8
        for (int k = 0; k < RNN; ++k) s += sh_h[k] * w[k];
        sh_pq[tid] = s;
    }
    __syncthreads();
    {   // per-t: conv(cum) -> ploc -> energy
        int t = tid;
        float loc[FILT];
        #pragma unroll
        for (int f = 0; f < FILT; ++f) {
            float s = conv_b[f];
            const float* w = conv_w + f * KSZ;
            #pragma unroll
            for (int k = 0; k < KSZ; ++k) s += sh_cum[t + k] * w[k];
            loc[f] = s;
        }
        const float* pmrow = pm + ((size_t)b * T_ENC + t) * ATTN_D;
        float e = 0.f;
        for (int d = 0; d < ATTN_D; ++d) {
            float pl = 0.f;
            const float* wl = WL + d * FILT;
            #pragma unroll
            for (int f = 0; f < FILT; ++f) pl += loc[f] * wl[f];
            e += vvec[d] * tanhf(sh_pq[d] + pmrow[d] + pl);
        }
        if (t >= mlen[b]) e = -1e9f;
        sh_e[t] = e;
    }
    __syncthreads();
    // softmax over 256
    sred[tid] = sh_e[tid];
    __syncthreads();
    for (int off = 128; off > 0; off >>= 1) {
        if (tid < off) sred[tid] = fmaxf(sred[tid], sred[tid + off]);
        __syncthreads();
    }
    float mx = sred[0];
    __syncthreads();
    float ex = __expf(sh_e[tid] - mx);
    sred[tid] = ex;
    __syncthreads();
    for (int off = 128; off > 0; off >>= 1) {
        if (tid < off) sred[tid] += sred[tid + off];
        __syncthreads();
    }
    float al = ex / sred[0];
    __syncthreads();
    sh_e[tid] = al;  // alignment
    cum[(size_t)b * T_ENC + tid] += al;
    out_attn[((size_t)b * NSTEPS + step) * T_ENC + tid] = al;
    __syncthreads();
    // ctx = align @ enc
    for (int d = tid; d < ENC_D; d += 256) {
        float s = 0.f;
        for (int t = 0; t < T_ENC; ++t)
            s += sh_e[t] * enc[((size_t)b * T_ENC + t) * ENC_D + d];
        sh_ctx[d] = s;
        ctx[(size_t)b * ENC_D + d] = s;
    }
    __syncthreads();
    // pack u2 = [attn_h(1024) | ctx(512) | dec_h(1024)] for the decoder GEMM
    int mt = b >> 4, lanebase = b & 15;
    const float* dhb = dec_h + (size_t)b * RNN;
    for (int idx = tid; idx < NKT_D * 16; idx += 256) {
        int kt = idx >> 4, khalf = (idx >> 3) & 1, v = idx & 7;
        int lane = lanebase + 16 * khalf;
        int k0 = kt * 32 + frag_k(v, khalf);
        float w0, w1;
        {
            int k = k0;
            w0 = (k < RNN) ? sh_h[k] : (k < RNN + ENC_D) ? sh_ctx[k - RNN] : dhb[k - RNN - ENC_D];
            k = k0 + 1;
            w1 = (k < RNN) ? sh_h[k] : (k < RNN + ENC_D) ? sh_ctx[k - RNN] : dhb[k - RNN - ENC_D];
        }
        unsigned short* dst = apack_d + (((size_t)(mt * NKT_D + kt) * 32 + lane) * 16 + 2 * v);
        dst[0] = f32_to_bf16(w0);
        dst[1] = f32_to_bf16(w1);
    }
}

// ---------------- S6: decoder LSTM cell + mel/stop projection (fused) --------
__global__ void dec_cell_proj_kernel(const float* __restrict__ gates,
                                     const float* __restrict__ bih, const float* __restrict__ bhh,
                                     float* __restrict__ h, float* __restrict__ c,
                                     const float* __restrict__ ctx,
                                     const float* __restrict__ Wmel, const float* __restrict__ bmel,
                                     const float* __restrict__ Wstop, const float* __restrict__ bstop,
                                     float* __restrict__ out_mel, float* __restrict__ out_stop,
                                     int step) {
    __shared__ float sh[PROJ_K];  // [dec_h(1024) | ctx(512)]
    __shared__ float sred[256];
    int b = blockIdx.x, tid = threadIdx.x;
    const float* g = gates + (size_t)b * NGATE;
    for (int j = tid; j < RNN; j += 256) {
        float gi = g[j]        + bih[j]        + bhh[j];
        float gf = g[j + 1024] + bih[j + 1024] + bhh[j + 1024];
        float gg = g[j + 2048] + bih[j + 2048] + bhh[j + 2048];
        float go = g[j + 3072] + bih[j + 3072] + bhh[j + 3072];
        float cc = sigmoidf_(gf) * c[(size_t)b * RNN + j] + sigmoidf_(gi) * tanhf(gg);
        float hh = sigmoidf_(go) * tanhf(cc);
        c[(size_t)b * RNN + j] = cc;
        h[(size_t)b * RNN + j] = hh;
        sh[j] = hh;
    }
    for (int j = tid; j < ENC_D; j += 256) sh[RNN + j] = ctx[(size_t)b * ENC_D + j];
    __syncthreads();
    if (tid < MEL * R_) {   // mel projection (160 outputs)
        float s = bmel[tid];
        const float* w = Wmel + (size_t)tid * PROJ_K;
        #pragma unroll 8
        for (int k = 0; k < PROJ_K; ++k) s += sh[k] * w[k];
        int r = tid / MEL, mcol = tid % MEL;
        out_mel[((size_t)b * T_DEC + step * R_ + r) * MEL + mcol] = s;
    }
    // stop token (block-wide dot product)
    float p = 0.f;
    for (int k = tid; k < PROJ_K; k += 256) p += sh[k] * Wstop[k];
    sred[tid] = p;
    __syncthreads();
    for (int off = 128; off > 0; off >>= 1) {
        if (tid < off) sred[tid] += sred[tid + off];
        __syncthreads();
    }
    if (tid == 0) {
        float st = sigmoidf_(sred[0] + bstop[0]);
        out_stop[(size_t)b * T_DEC + 2 * step]     = st;
        out_stop[(size_t)b * T_DEC + 2 * step + 1] = st;
    }
}

// ---------------- host-side launch ----------------
extern "C" void kernel_launch(void* const* d_in, const int* in_sizes, int n_in,
                              void* d_out, int out_size, void* d_ws, size_t ws_size,
                              hipStream_t stream) {
    const float* enc    = (const float*)d_in[0];
    const float* inputs = (const float*)d_in[1];
    const int*   mlen   = (const int*)d_in[2];
    const float* W1     = (const float*)d_in[3];
    const float* b1     = (const float*)d_in[4];
    const float* W2     = (const float*)d_in[5];
    const float* b2     = (const float*)d_in[6];
    const float* Wm     = (const float*)d_in[7];
    const float* Wih_a  = (const float*)d_in[8];
    const float* Whh_a  = (const float*)d_in[9];
    const float* bih_a  = (const float*)d_in[10];
    const float* bhh_a  = (const float*)d_in[11];
    const float* Wq     = (const float*)d_in[12];
    const float* conv_w = (const float*)d_in[13];
    const float* conv_b = (const float*)d_in[14];
    const float* WL     = (const float*)d_in[15];
    const float* vv     = (const float*)d_in[16];
    const float* Wih_d  = (const float*)d_in[17];
    const float* Whh_d  = (const float*)d_in[18];
    const float* bih_d  = (const float*)d_in[19];
    const float* bhh_d  = (const float*)d_in[20];
    const float* Wmel   = (const float*)d_in[21];
    const float* bmel   = (const float*)d_in[22];
    const float* Wstop  = (const float*)d_in[23];
    const float* bstop  = (const float*)d_in[24];

    char* ws = (char*)d_ws;
    unsigned short* wa_pack = (unsigned short*)(ws + OFF_WA);
    unsigned short* wd_pack = (unsigned short*)(ws + OFF_WD);
    float*          pm      = (float*)(ws + OFF_PM);
    unsigned short* apack_a = (unsigned short*)(ws + OFF_AA);
    unsigned short* apack_d = (unsigned short*)(ws + OFF_AD);
    float*          gates   = (float*)(ws + OFF_G);
    float*          state   = (float*)(ws + OFF_STATE);
    float* attn_h = state;
    float* attn_c = attn_h + B_ * RNN;
    float* dec_h  = attn_c + B_ * RNN;
    float* dec_c  = dec_h + B_ * RNN;
    float* ctx    = dec_c + B_ * RNN;
    float* cum    = ctx + B_ * ENC_D;

    float* out_mel  = (float*)d_out;                        // (32, 500, 80)
    float* out_stop = out_mel + (size_t)B_ * T_DEC * MEL;   // (32, 500)
    float* out_attn = out_stop + (size_t)B_ * T_DEC;        // (32, 250, 256)

    const size_t lds_a = (size_t)2 * NKT_A * 512 * sizeof(unsigned short); // 114,688 B
    const size_t lds_d = (size_t)2 * NKT_D * 512 * sizeof(unsigned short); // 163,840 B

    // ---- one-time-per-call init (deterministic; inside graph) ----
    {
        int tot_a = 256 * NKT_A * 32 * 8;
        pack_w_kernel<<<(tot_a + 255) / 256, 256, 0, stream>>>(Wih_a, PRE + ENC_D, Whh_a, RNN,
                                                               wa_pack, NKT_A);
        int tot_d = 256 * NKT_D * 32 * 8;
        pack_w_kernel<<<(tot_d + 255) / 256, 256, 0, stream>>>(Wih_d, RNN + ENC_D, Whh_d, RNN,
                                                               wd_pack, NKT_D);
        int tot_pm = B_ * T_ENC * ATTN_D;
        pm_kernel<<<(tot_pm + 255) / 256, 256, 0, stream>>>(enc, Wm, pm);
        zero_kernel<<<(N_STATE + 255) / 256, 256, 0, stream>>>(state, N_STATE);
    }

    // ---- 250 serial decoder steps, 6 kernels each ----
    for (int s = 0; s < NSTEPS; ++s) {
        prenet_pack_kernel<<<B_, 256, 0, stream>>>(inputs, W1, b1, W2, b2,
                                                   ctx, attn_h, apack_a, s);
        gemm_wmma_kernel<<<32, 256, lds_a, stream>>>(apack_a, wa_pack, gates, NKT_A);
        lstm_cell_kernel<<<B_, 256, 0, stream>>>(gates, bih_a, bhh_a, attn_h, attn_c);
        attention_kernel<<<B_, 256, 0, stream>>>(enc, pm, mlen, Wq, conv_w, conv_b, WL, vv,
                                                 attn_h, dec_h, ctx, cum, out_attn, apack_d, s);
        gemm_wmma_kernel<<<32, 256, lds_d, stream>>>(apack_d, wd_pack, gates, NKT_D);
        dec_cell_proj_kernel<<<B_, 256, 0, stream>>>(gates, bih_d, bhh_d, dec_h, dec_c, ctx,
                                                     Wmel, bmel, Wstop, bstop,
                                                     out_mel, out_stop, s);
    }
}
